// Block_18820546691143
// MI455X (gfx1250) — matrix operand order repaired and checked by reading, not compiled, not run on hardware
//
#include <hip/hip_runtime.h>

// ---------------------------------------------------------------------------
// GPT-2 style transformer block for MI455X (gfx1250, wave32, WMMA bf16).
// Pipeline: [weights->bf16T] -> LN1 -> QKV GEMM -> flash attention ->
//           proj GEMM (+bias+resid) -> LN2 -> FF1 (+bias+ReLU) -> FF2 (+bias+resid)
// GEMMs: v_wmma_f32_16x16x32_bf16, double-buffered LDS staging filled with
// global_load_async_to_lds_b128 (ASYNCcnt), 64x64 wave tiles.
// ---------------------------------------------------------------------------

typedef __bf16 bf16_t;
typedef __attribute__((ext_vector_type(16))) __bf16 v16bf;
typedef __attribute__((ext_vector_type(8)))  float  v8f;

constexpr int Bn   = 2;
constexpr int Tn   = 2048;
constexpr int Cn   = 768;
constexpr int Hn   = 12;
constexpr int Dn   = 64;
constexpr int Mn   = Bn * Tn;   // 4096 tokens
constexpr int FFn  = 4 * Cn;    // 3072
constexpr int NQKV = 3 * Cn;    // 2304

union FragU { uint4 u[2]; v16bf v; };

static __device__ __forceinline__ v16bf ldfrag(const bf16_t* p0, const bf16_t* p1) {
  FragU t;
  t.u[0] = *reinterpret_cast<const uint4*>(p0);
  t.u[1] = *reinterpret_cast<const uint4*>(p1);
  return t.v;
}

static __device__ __forceinline__ v8f wmma_bf16(v16bf a, v16bf b, v8f c) {
  // (neg_a, A, neg_b, B, c_mod, C, reuse_a, reuse_b)
  return __builtin_amdgcn_wmma_f32_16x16x32_bf16(false, a, false, b, (short)0, c, false, false);
}

// async 16B global -> LDS copy (ASYNCcnt tracked).  LDS address = low 32 bits
// of the generic pointer (ISA 10.2).
static __device__ __forceinline__ void cp_async16(bf16_t* ldst, const bf16_t* gsrc) {
  asm volatile("global_load_async_to_lds_b128 %0, %1, off"
               :: "v"((unsigned)(unsigned long long)(void*)ldst),
                  "v"((unsigned long long)(const void*)gsrc)
               : "memory");
}
static __device__ __forceinline__ void wait_async0() {
  asm volatile("s_wait_asynccnt 0x0" ::: "memory");
}

// ---------------- weight prep: f32 [K,N] -> bf16 [N,K] ----------------------
__global__ void k_transpose_w(const float* __restrict__ src, bf16_t* __restrict__ dst,
                              int K, int N) {
  int idx = blockIdx.x * 256 + threadIdx.x;        // over N*K (exact)
  int n = idx / K, kk = idx - n * K;
  dst[idx] = (bf16_t)src[(size_t)kk * N + n];
}

// Wq/Wk/Wv [H,C,D] -> WtQKV [2304][768]  (n = part*768 + h*64 + d, k = c)
__global__ void k_qkv_w(const float* __restrict__ Wq, const float* __restrict__ Wk,
                        const float* __restrict__ Wv, bf16_t* __restrict__ dst) {
  int idx = blockIdx.x * 256 + threadIdx.x;        // over 2304*768
  int n = idx / Cn, c = idx - n * Cn;
  int part = n / Cn; int r = n - part * Cn; int h = r >> 6; int d = r & 63;
  const float* W = (part == 0) ? Wq : (part == 1 ? Wk : Wv);
  dst[idx] = (bf16_t)W[((size_t)h * Cn + c) * Dn + d];
}

// ---------------- layernorm: f32 in -> bf16 out -----------------------------
__global__ __launch_bounds__(256) void k_ln(const float* __restrict__ x,
                                            const float* __restrict__ g,
                                            const float* __restrict__ be,
                                            bf16_t* __restrict__ out) {
  __shared__ float r1[256], r2[256];
  int row = blockIdx.x, tid = threadIdx.x;
  const float* xr = x + (size_t)row * Cn;
  float v[3], s = 0.f, s2 = 0.f;
#pragma unroll
  for (int i = 0; i < 3; ++i) { float t = xr[tid + i * 256]; v[i] = t; s += t; s2 += t * t; }
  r1[tid] = s; r2[tid] = s2; __syncthreads();
  for (int off = 128; off > 0; off >>= 1) {
    if (tid < off) { r1[tid] += r1[tid + off]; r2[tid] += r2[tid + off]; }
    __syncthreads();
  }
  float mu = r1[0] * (1.0f / Cn);
  float var = r2[0] * (1.0f / Cn) - mu * mu;
  float rs = rsqrtf(var + 1e-5f);
  bf16_t* orow = out + (size_t)row * Cn;
#pragma unroll
  for (int i = 0; i < 3; ++i) { int c = tid + i * 256; orow[c] = (bf16_t)((v[i] - mu) * rs * g[c] + be[c]); }
}

// ---------------- GEMM: C[M,N] = A[M,K] * Bt[N,K]^T -------------------------
// block 128 thr = 4 waves (2x2); block tile 128x128, wave tile 64x64.
// K staged 32 deep, double buffered, filled with async global->LDS copies.
// EPI 0: qkv split store (q,k [B,H,T,D] bf16; v transposed [B,H,D,T] bf16)
// EPI 1: bias + ReLU -> bf16 [M,N]
// EPI 2: bias + resid -> f32 [M,N]
template<int EPI>
__global__ __launch_bounds__(128) void k_gemm(const bf16_t* __restrict__ A,
                                              const bf16_t* __restrict__ Bt,
                                              const float* __restrict__ bias,
                                              const float* __restrict__ resid,
                                              float* __restrict__ outf,
                                              bf16_t* __restrict__ ob,
                                              bf16_t* __restrict__ o1,
                                              bf16_t* __restrict__ o2,
                                              int N, int K) {
  __shared__ alignas(16) bf16_t As[2][128 * 40];   // 128 rows x 32 k (+8 pad)
  __shared__ alignas(16) bf16_t Bs[2][128 * 40];
  int tid = threadIdx.x, wid = tid >> 5, lane = tid & 31, nl = lane & 15, hi = lane >> 4;
  int m0 = blockIdx.y * 128, n0 = blockIdx.x * 128;
  int wr = (wid >> 1) * 64, wc = (wid & 1) * 64;
  v8f acc[4][4] = {};

  auto issue = [&](int k0, int buf) {
#pragma unroll
    for (int i = 0; i < 4; ++i) {                  // 512 chunks A + 512 chunks B
      int c = i * 128 + tid; int rr = c >> 2, cc = (c & 3) * 8;
      cp_async16(&As[buf][rr * 40 + cc], &A[(size_t)(m0 + rr) * K + k0 + cc]);
      cp_async16(&Bs[buf][rr * 40 + cc], &Bt[(size_t)(n0 + rr) * K + k0 + cc]);
    }
  };

  int nst = K / 32;
  issue(0, 0);
  for (int s = 0; s < nst; ++s) {
    wait_async0();
    __syncthreads();                               // stage s visible to all waves
    if (s + 1 < nst) issue((s + 1) * 32, (s + 1) & 1);
    const bf16_t* Ab = &As[s & 1][0];
    const bf16_t* Bb = &Bs[s & 1][0];
    v16bf af[4];
#pragma unroll
    for (int m4 = 0; m4 < 4; ++m4) {
      const bf16_t* ar = &Ab[(wr + m4 * 16 + nl) * 40 + hi * 8];
      af[m4] = ldfrag(ar, ar + 16);
    }
#pragma unroll
    for (int nn = 0; nn < 4; ++nn) {
      const bf16_t* br = &Bb[(wc + nn * 16 + nl) * 40 + hi * 8];
      v16bf bfm = ldfrag(br, br + 16);
#pragma unroll
      for (int m4 = 0; m4 < 4; ++m4) acc[m4][nn] = wmma_bf16(af[m4], bfm, acc[m4][nn]);
    }
    __syncthreads();                               // compute done before reuse of buf
  }

  // epilogue: C-layout element e -> row = e + 8*hi, col = nl
#pragma unroll
  for (int m4 = 0; m4 < 4; ++m4)
#pragma unroll
    for (int nn = 0; nn < 4; ++nn)
#pragma unroll
      for (int e = 0; e < 8; ++e) {
        int mm  = m0 + wr + m4 * 16 + e + hi * 8;
        int nng = n0 + wc + nn * 16 + nl;
        float val = acc[m4][nn][e];
        if constexpr (EPI == 0) {
          int part = nng / Cn, r = nng - part * Cn, h = r >> 6, d = r & 63;
          int b = mm >> 11, t = mm & (Tn - 1);
          size_t bh = (size_t)(b * Hn + h);
          if (part == 0)      ob[(bh * Tn + t) * Dn + d] = (bf16_t)val;
          else if (part == 1) o1[(bh * Tn + t) * Dn + d] = (bf16_t)val;
          else                o2[(bh * Dn + d) * Tn + t] = (bf16_t)val;   // V^T
        } else if constexpr (EPI == 1) {
          val += bias[nng]; val = fmaxf(val, 0.0f);
          ob[(size_t)mm * N + nng] = (bf16_t)val;
        } else {
          val += bias[nng] + resid[(size_t)mm * N + nng];
          outf[(size_t)mm * N + nng] = val;
        }
      }
}

// ---------------- causal flash attention ------------------------------------
// one wave per 16-row Q block; streams 64 keys per step; online softmax.
__global__ __launch_bounds__(128) void k_attn(const bf16_t* __restrict__ Q,
                                              const bf16_t* __restrict__ Kb,
                                              const bf16_t* __restrict__ Vt,
                                              bf16_t* __restrict__ out) {
  __shared__ alignas(16) bf16_t P[4][16 * 64];
  int tid = threadIdx.x, wid = tid >> 5, lane = tid & 31, nl = lane & 15, hi = lane >> 4;
  int task = blockIdx.x * 4 + wid;          // 0..3071
  int qb = task & 127, bh = task >> 7;      // qb: 16-row block, bh: b*H+h
  int t0 = qb * 16;
  const float scale = 0.03608439182435161f; // 768^-0.5 (faithful: scales by C^-0.5)
  const bf16_t* qrow = Q + ((size_t)bh * Tn + t0 + nl) * Dn;
  v16bf qa0 = ldfrag(qrow + hi * 8,      qrow + hi * 8 + 16);
  v16bf qa1 = ldfrag(qrow + 32 + hi * 8, qrow + 32 + hi * 8 + 16);
  const bf16_t* kbase = Kb + (size_t)bh * Tn * Dn;
  const bf16_t* vbase = Vt + (size_t)bh * Dn * Tn;
  bf16_t* Pw = &P[wid][0];
  float rmax[8], rsum[8];
  v8f Oc[4] = {};
#pragma unroll
  for (int e = 0; e < 8; ++e) { rmax[e] = -__builtin_inff(); rsum[e] = 0.0f; }
  int jend = (t0 + 15) >> 6;                // 64-key blocks
  for (int j = 0; j <= jend; ++j) {
    int s0 = j * 64;
    v8f S[4] = {};
#pragma unroll
    for (int sub = 0; sub < 4; ++sub) {
      const bf16_t* kr = kbase + (size_t)(s0 + sub * 16 + nl) * Dn;
      S[sub] = wmma_bf16(qa0, ldfrag(kr + hi * 8,      kr + hi * 8 + 16),      S[sub]);
      S[sub] = wmma_bf16(qa1, ldfrag(kr + 32 + hi * 8, kr + 32 + hi * 8 + 16), S[sub]);
    }
    float tmax[8];
#pragma unroll
    for (int e = 0; e < 8; ++e) {
      float v0 = S[0][e] * scale, v1 = S[1][e] * scale;
      float v2 = S[2][e] * scale, v3 = S[3][e] * scale;
      if (j == jend) {                      // diagonal block: causal mask
        int tr = t0 + e + hi * 8;
        if (s0 +      nl > tr) v0 = -__builtin_inff();
        if (s0 + 16 + nl > tr) v1 = -__builtin_inff();
        if (s0 + 32 + nl > tr) v2 = -__builtin_inff();
        if (s0 + 48 + nl > tr) v3 = -__builtin_inff();
      }
      S[0][e] = v0; S[1][e] = v1; S[2][e] = v2; S[3][e] = v3;
      float mx = fmaxf(fmaxf(v0, v1), fmaxf(v2, v3));
#pragma unroll
      for (int off = 1; off < 16; off <<= 1) mx = fmaxf(mx, __shfl_xor(mx, off, 32));
      tmax[e] = mx;                         // row max across 16 lanes (half-wave)
    }
    float al[8], ps[8];
#pragma unroll
    for (int e = 0; e < 8; ++e) {
      float nm = fmaxf(rmax[e], tmax[e]);
      al[e] = __expf(rmax[e] - nm);
      rmax[e] = nm;
      int row = e + hi * 8;
      float pp = 0.f;
#pragma unroll
      for (int sub = 0; sub < 4; ++sub) {
        float p = __expf(S[sub][e] - nm);
        Pw[row * 64 + sub * 16 + nl] = (bf16_t)p;   // stage P for A-frag reload
        pp += p;
      }
#pragma unroll
      for (int off = 1; off < 16; off <<= 1) pp += __shfl_xor(pp, off, 32);
      ps[e] = pp;
    }
#pragma unroll
    for (int e = 0; e < 8; ++e) rsum[e] = rsum[e] * al[e] + ps[e];
#pragma unroll
    for (int c = 0; c < 4; ++c)
#pragma unroll
      for (int e = 0; e < 8; ++e) Oc[c][e] *= al[e];
#pragma unroll
    for (int kk = 0; kk < 2; ++kk) {        // P*V over the 64 keys (2 k-steps)
      const bf16_t* prow = Pw + nl * 64 + kk * 32 + hi * 8;
      v16bf pf = ldfrag(prow, prow + 16);
#pragma unroll
      for (int c = 0; c < 4; ++c) {
        const bf16_t* vr = vbase + (size_t)(c * 16 + nl) * Tn + s0 + kk * 32 + hi * 8;
        Oc[c] = wmma_bf16(pf, ldfrag(vr, vr + 16), Oc[c]);
      }
    }
  }
  int bb = bh / Hn, hh = bh - bb * Hn;
#pragma unroll
  for (int e = 0; e < 8; ++e) {
    float inv = 1.0f / rsum[e];
    size_t rowoff = (size_t)(bb * Tn + t0 + e + hi * 8) * Cn + hh * Dn;
#pragma unroll
    for (int c = 0; c < 4; ++c) out[rowoff + c * 16 + nl] = (bf16_t)(Oc[c][e] * inv);
  }
}

// ---------------------------------------------------------------------------
extern "C" void kernel_launch(void* const* d_in, const int* in_sizes, int n_in,
                              void* d_out, int out_size, void* d_ws, size_t ws_size,
                              hipStream_t stream) {
  (void)in_sizes; (void)n_in; (void)out_size; (void)ws_size;
  const float* x      = (const float*)d_in[0];
  const float* Wq     = (const float*)d_in[1];
  const float* Wk     = (const float*)d_in[2];
  const float* Wv     = (const float*)d_in[3];
  const float* Wproj  = (const float*)d_in[4];
  const float* bproj  = (const float*)d_in[5];
  const float* Wff1   = (const float*)d_in[6];
  const float* bff1   = (const float*)d_in[7];
  const float* Wff2   = (const float*)d_in[8];
  const float* bff2   = (const float*)d_in[9];
  const float* g1     = (const float*)d_in[10];
  const float* beta1  = (const float*)d_in[11];
  const float* g2     = (const float*)d_in[12];
  const float* beta2  = (const float*)d_in[13];

  char* w = (char*)d_ws;
  auto take = [&](size_t sz) { char* p = w; w += sz; return p; };
  bf16_t* WtQKV  = (bf16_t*)take((size_t)NQKV * Cn * 2);   // [2304][768]
  bf16_t* WtProj = (bf16_t*)take((size_t)Cn * Cn * 2);     // [768][768]
  bf16_t* WtFF1  = (bf16_t*)take((size_t)FFn * Cn * 2);    // [3072][768]
  bf16_t* WtFF2  = (bf16_t*)take((size_t)Cn * FFn * 2);    // [768][3072]
  bf16_t* hb     = (bf16_t*)take((size_t)Mn * Cn * 2);     // LN1 out
  bf16_t* qbuf   = (bf16_t*)take((size_t)Mn * Cn * 2);     // [B,H,T,D]
  bf16_t* kbuf   = (bf16_t*)take((size_t)Mn * Cn * 2);     // [B,H,T,D]
  bf16_t* vtbuf  = (bf16_t*)take((size_t)Mn * Cn * 2);     // [B,H,D,T]
  bf16_t* attb   = (bf16_t*)take((size_t)Mn * Cn * 2);     // concat heads [M,C]
  float*  x1     = (float*) take((size_t)Mn * Cn * 4);     // after proj + resid
  bf16_t* h2b    = (bf16_t*)take((size_t)Mn * Cn * 2);     // LN2 out
  bf16_t* ffb    = (bf16_t*)take((size_t)Mn * FFn * 2);    // relu(ff1)
  float*  outf   = (float*)d_out;

  // weight conversion (one pass, ~28 MB)
  k_qkv_w      <<<(NQKV * Cn) / 256, 256, 0, stream>>>(Wq, Wk, Wv, WtQKV);
  k_transpose_w<<<(Cn * Cn)  / 256, 256, 0, stream>>>(Wproj, WtProj, Cn, Cn);
  k_transpose_w<<<(Cn * FFn) / 256, 256, 0, stream>>>(Wff1,  WtFF1,  Cn, FFn);
  k_transpose_w<<<(Cn * FFn) / 256, 256, 0, stream>>>(Wff2,  WtFF2,  FFn, Cn);

  // LN1
  k_ln<<<Mn, 256, 0, stream>>>(x, g1, beta1, hb);
  // QKV projections
  k_gemm<0><<<dim3(NQKV / 128, Mn / 128), 128, 0, stream>>>(
      hb, WtQKV, nullptr, nullptr, nullptr, qbuf, kbuf, vtbuf, NQKV, Cn);
  // attention (24 heads * 128 q-blocks / 4 waves per block)
  k_attn<<<(Bn * Hn * (Tn / 16)) / 4, 128, 0, stream>>>(qbuf, kbuf, vtbuf, attb);
  // output projection + bias + residual(x)
  k_gemm<2><<<dim3(Cn / 128, Mn / 128), 128, 0, stream>>>(
      attb, WtProj, bproj, x, x1, nullptr, nullptr, nullptr, Cn, Cn);
  // LN2
  k_ln<<<Mn, 256, 0, stream>>>(x1, g2, beta2, h2b);
  // FF1 + bias + ReLU
  k_gemm<1><<<dim3(FFn / 128, Mn / 128), 128, 0, stream>>>(
      h2b, WtFF1, bff1, nullptr, nullptr, ffb, nullptr, nullptr, FFn, Cn);
  // FF2 + bias + residual(x1) -> out
  k_gemm<2><<<dim3(Cn / 128, Mn / 128), 128, 0, stream>>>(
      ffb, WtFF2, bff2, x1, outf, nullptr, nullptr, nullptr, Cn, FFn);
}
